// SAModule2_86921548136901
// MI455X (gfx1250) — matrix operand order repaired and checked by reading, not compiled, not run on hardware
//
#include <hip/hip_runtime.h>
#include <hip/hip_bf16.h>

// ---------------------------------------------------------------------------
// PointNet++ SA module for MI455X (gfx1250, wave32, WMMA)
//   B=4 clouds, N=8192 pts, S=2048 centroids, K=64 nbrs, MLP 67->64->64->128
// Pipeline: fps -> ball query -> fused gather+MLP(WMMA f16)+max -> tail
// ---------------------------------------------------------------------------

#define BB   4
#define NN   8192
#define SS   2048
#define KK   64
#define FIN  64
#define H1D  64
#define H2D  64
#define FOUT 128
#define R2   0.04f
#define KPAD 96          // 67 padded to 3 x 32 K-chunks

typedef _Float16 h16;
typedef __attribute__((ext_vector_type(16))) _Float16 v16h;
typedef __attribute__((ext_vector_type(8)))  float    v8f;

union Frag16 { v16h v; unsigned u[8]; };

// ----------------------------- FPS -----------------------------------------
__device__ __forceinline__ void argmax_combine(float& v, int& i, float v2, int i2) {
    if (v2 > v || (v2 == v && i2 < i)) { v = v2; i = i2; }
}

__global__ __launch_bounds__(1024)
void fps_kernel(const float* __restrict__ pos, int* __restrict__ fps_idx,
                float* __restrict__ qpos) {
    const int b = blockIdx.x, tid = threadIdx.x;
    const float* p = pos + (size_t)b * NN * 3;
    const int lane = tid & 31, wv = tid >> 5;

    float px[8], py[8], pz[8], d[8];
    const float c0x = p[0], c0y = p[1], c0z = p[2];
#pragma unroll
    for (int j = 0; j < 8; ++j) {
        const int i = j * 1024 + tid;
        px[j] = p[i * 3 + 0]; py[j] = p[i * 3 + 1]; pz[j] = p[i * 3 + 2];
        const float dx = px[j] - c0x, dy = py[j] - c0y, dz = pz[j] - c0z;
        d[j] = dx * dx + dy * dy + dz * dz;
    }
    if (tid == 0) {
        fps_idx[b * SS] = 0;
        qpos[(size_t)(b * SS) * 3 + 0] = c0x;
        qpos[(size_t)(b * SS) * 3 + 1] = c0y;
        qpos[(size_t)(b * SS) * 3 + 2] = c0z;
    }

    __shared__ float rv[32];
    __shared__ int   ri[32];
    __shared__ float cur[4];

    for (int s = 1; s < SS; ++s) {
        float bv = d[0]; int bi = tid;
#pragma unroll
        for (int j = 1; j < 8; ++j) argmax_combine(bv, bi, d[j], j * 1024 + tid);
#pragma unroll
        for (int off = 16; off >= 1; off >>= 1) {
            float ov = __shfl_xor(bv, off, 32);
            int   oi = __shfl_xor(bi, off, 32);
            argmax_combine(bv, bi, ov, oi);
        }
        if (lane == 0) { rv[wv] = bv; ri[wv] = bi; }
        __syncthreads();
        if (wv == 0) {
            float v = rv[lane]; int i = ri[lane];
#pragma unroll
            for (int off = 16; off >= 1; off >>= 1) {
                float ov = __shfl_xor(v, off, 32);
                int   oi = __shfl_xor(i, off, 32);
                argmax_combine(v, i, ov, oi);
            }
            if (lane == 0) {
                const int win = i;
                fps_idx[b * SS + s] = win;
                const float wx = p[win * 3 + 0], wy = p[win * 3 + 1], wz = p[win * 3 + 2];
                cur[0] = wx; cur[1] = wy; cur[2] = wz;
                qpos[(size_t)(b * SS + s) * 3 + 0] = wx;
                qpos[(size_t)(b * SS + s) * 3 + 1] = wy;
                qpos[(size_t)(b * SS + s) * 3 + 2] = wz;
            }
        }
        __syncthreads();
        const float cx = cur[0], cy = cur[1], cz = cur[2];
#pragma unroll
        for (int j = 0; j < 8; ++j) {
            const float dx = px[j] - cx, dy = py[j] - cy, dz = pz[j] - cz;
            d[j] = fminf(d[j], dx * dx + dy * dy + dz * dz);
        }
    }
}

// ------------------------- radius ball query --------------------------------
__global__ __launch_bounds__(256)
void ballq_kernel(const float* __restrict__ pos, const float* __restrict__ qpos,
                  const int* __restrict__ fps_idx, int* __restrict__ nbr,
                  int* __restrict__ cnts) {
    const int gw = blockIdx.x * 8 + (threadIdx.x >> 5);   // one wave per centroid
    const int lane = threadIdx.x & 31;
    if (gw >= BB * SS) return;
    const int b = gw / SS;
    const float* p = pos + (size_t)b * NN * 3;
    const float qx = qpos[(size_t)gw * 3 + 0];
    const float qy = qpos[(size_t)gw * 3 + 1];
    const float qz = qpos[(size_t)gw * 3 + 2];

    int cnt = 0;
    for (int base = 0; base < NN && cnt < KK; base += 32) {
        const int i = base + lane;
        const float dx = p[i * 3 + 0] - qx, dy = p[i * 3 + 1] - qy, dz = p[i * 3 + 2] - qz;
        const bool in = (dx * dx + dy * dy + dz * dz) <= R2;
        const unsigned m = (unsigned)__ballot(in);
        const int pre = __popc(m & ((1u << lane) - 1u));
        const int slot = cnt + pre;
        if (in && slot < KK) nbr[(size_t)gw * KK + slot] = i;
        cnt += __popc(m);
    }
    if (cnt > KK) cnt = KK;
    const int ctr = fps_idx[gw];                  // centroid idx (always in-range)
    for (int k = cnt + lane; k < KK; k += 32) nbr[(size_t)gw * KK + k] = ctr;
    if (lane == 0) cnts[gw] = cnt;
}

// -------------------- weight transpose+convert to f16 -----------------------
__global__ __launch_bounds__(256)
void wconv_kernel(const float* __restrict__ W1, const float* __restrict__ W2,
                  const float* __restrict__ W3, h16* __restrict__ w1t,
                  h16* __restrict__ w2t, h16* __restrict__ w3t) {
    const int t = threadIdx.x;
    for (int i = t; i < H1D * KPAD; i += 256) {           // [64 cols][96 k]
        const int col = i / KPAD, k = i % KPAD;
        w1t[i] = (h16)((k < FIN + 3) ? W1[k * H1D + col] : 0.f);
    }
    for (int i = t; i < H2D * H1D; i += 256) {            // [64 cols][64 k]
        const int col = i / H1D, k = i % H1D;
        w2t[i] = (h16)W2[k * H2D + col];
    }
    for (int i = t; i < FOUT * H2D; i += 256) {           // [128 cols][64 k]
        const int col = i / H2D, k = i % H2D;
        w3t[i] = (h16)W3[k * FOUT + col];
    }
}

// -------------------- WMMA fragment loaders ---------------------------------
// A (16x32 f16): lanes 0-15 M=row, v0..3 -> K=2v,2v+1 ; v4..7 -> K=16+2(v-4)..
//                lanes 16-31 same rows, K offset +8.
__device__ __forceinline__ v16h load_a_lds(const h16* base, int stride, int row,
                                           int kbase, int lane) {
    Frag16 f;
    const int hi = lane >> 4;
    const h16* p = base + row * stride + kbase + hi * 8;
#pragma unroll
    for (int v = 0; v < 4; ++v) f.u[v] = *(const unsigned*)(p + 2 * v);
#pragma unroll
    for (int v = 4; v < 8; ++v) f.u[v] = *(const unsigned*)(p + 16 + 2 * (v - 4));
    return f.v;
}

// B (32x16 f16), weights stored col-major [col][kstride]:
//   lanes 0-15: col = lane, K=0..7,16..23 ; lanes 16-31: same col set, K=8..15,24..31.
__device__ __forceinline__ v16h load_b_glb(const h16* w, int kstride, int col,
                                           int kbase, int lane) {
    Frag16 f;
    const int hi = lane >> 4;
    const h16* p = w + col * kstride + kbase + hi * 8;
#pragma unroll
    for (int v = 0; v < 4; ++v) f.u[v] = *(const unsigned*)(p + 2 * v);
#pragma unroll
    for (int v = 4; v < 8; ++v) f.u[v] = *(const unsigned*)(p + 16 + 2 * (v - 4));
    return f.v;
}

// ------------- fused gather -> MLP (WMMA) -> masked max ---------------------
__global__ __launch_bounds__(128)
void mlp_kernel(const float* __restrict__ x, const float* __restrict__ pos,
                const int* __restrict__ nbr, const int* __restrict__ cnts,
                const float* __restrict__ qpos,
                const h16* __restrict__ w1t, const h16* __restrict__ w2t,
                const h16* __restrict__ w3t,
                const float* __restrict__ b1, const float* __restrict__ b2,
                const float* __restrict__ b3, float* __restrict__ out_x) {
    const int c = blockIdx.x, tid = threadIdx.x;
    const int b = c / SS;

    __shared__ h16  hA[64 * KPAD];   // gathered [x_j, p_j - q] rows, f16
    __shared__ h16  h1s[64 * 64];
    __shared__ h16  h2s[64 * 64];
    __shared__ float pmax[4 * FOUT];
    __shared__ int  snbr[64];

    if (tid < 64) snbr[tid] = nbr[(size_t)c * KK + tid];
    __syncthreads();

    { // gather features: 2 threads per row, 32 feats each, float4 vectorized
        const int r = tid >> 1, fh = (tid & 1) * 32;
        const float4* src = (const float4*)(x + ((size_t)b * NN + snbr[r]) * FIN + fh);
        h16* dst = hA + r * KPAD + fh;
#pragma unroll
        for (int q = 0; q < 8; ++q) {
            const float4 t = src[q];
            dst[4 * q + 0] = (h16)t.x; dst[4 * q + 1] = (h16)t.y;
            dst[4 * q + 2] = (h16)t.z; dst[4 * q + 3] = (h16)t.w;
        }
    }
    if (tid < 64) { // relative position + zero pad to KPAD
        const int r = tid;
        const float* pp = pos + ((size_t)b * NN + snbr[r]) * 3;
        const float qx = qpos[(size_t)c * 3 + 0];
        const float qy = qpos[(size_t)c * 3 + 1];
        const float qz = qpos[(size_t)c * 3 + 2];
        hA[r * KPAD + 64] = (h16)(pp[0] - qx);
        hA[r * KPAD + 65] = (h16)(pp[1] - qy);
        hA[r * KPAD + 66] = (h16)(pp[2] - qz);
#pragma unroll
        for (int k = 67; k < KPAD; ++k) hA[r * KPAD + k] = (h16)0.f;
    }
    __syncthreads();

    const int lane = tid & 31, w = tid >> 5;
    const int m0 = w * 16;
    const int row = lane & 15, hi = lane >> 4, col16 = lane & 15;
    const int count = cnts[c];

    // ---- Layer 1: [64x96] x [96x64] ----
    {
        const v16h a0 = load_a_lds(hA, KPAD, m0 + row, 0,  lane);
        const v16h a1 = load_a_lds(hA, KPAD, m0 + row, 32, lane);
        const v16h a2 = load_a_lds(hA, KPAD, m0 + row, 64, lane);
#pragma unroll
        for (int n = 0; n < 4; ++n) {
            v8f acc = {};
            acc = __builtin_amdgcn_wmma_f32_16x16x32_f16(false, a0, false,
                  load_b_glb(w1t, KPAD, n * 16 + col16, 0,  lane), (short)0, acc, false, false);
            acc = __builtin_amdgcn_wmma_f32_16x16x32_f16(false, a1, false,
                  load_b_glb(w1t, KPAD, n * 16 + col16, 32, lane), (short)0, acc, false, false);
            acc = __builtin_amdgcn_wmma_f32_16x16x32_f16(false, a2, false,
                  load_b_glb(w1t, KPAD, n * 16 + col16, 64, lane), (short)0, acc, false, false);
            const float bias = b1[n * 16 + col16];
#pragma unroll
            for (int r = 0; r < 8; ++r) {
                const float v = fmaxf(acc[r] + bias, 0.f);
                h1s[(m0 + r + 8 * hi) * 64 + n * 16 + col16] = (h16)v;
            }
        }
    }
    __syncthreads();

    // ---- Layer 2: [64x64] x [64x64] ----
    {
        const v16h a0 = load_a_lds(h1s, 64, m0 + row, 0,  lane);
        const v16h a1 = load_a_lds(h1s, 64, m0 + row, 32, lane);
#pragma unroll
        for (int n = 0; n < 4; ++n) {
            v8f acc = {};
            acc = __builtin_amdgcn_wmma_f32_16x16x32_f16(false, a0, false,
                  load_b_glb(w2t, 64, n * 16 + col16, 0,  lane), (short)0, acc, false, false);
            acc = __builtin_amdgcn_wmma_f32_16x16x32_f16(false, a1, false,
                  load_b_glb(w2t, 64, n * 16 + col16, 32, lane), (short)0, acc, false, false);
            const float bias = b2[n * 16 + col16];
#pragma unroll
            for (int r = 0; r < 8; ++r) {
                const float v = fmaxf(acc[r] + bias, 0.f);
                h2s[(m0 + r + 8 * hi) * 64 + n * 16 + col16] = (h16)v;
            }
        }
    }
    __syncthreads();

    // ---- Layer 3: [64x64] x [64x128] + validity mask + per-wave max ----
    {
        const v16h a0 = load_a_lds(h2s, 64, m0 + row, 0,  lane);
        const v16h a1 = load_a_lds(h2s, 64, m0 + row, 32, lane);
#pragma unroll
        for (int n = 0; n < 8; ++n) {
            v8f acc = {};
            acc = __builtin_amdgcn_wmma_f32_16x16x32_f16(false, a0, false,
                  load_b_glb(w3t, 64, n * 16 + col16, 0,  lane), (short)0, acc, false, false);
            acc = __builtin_amdgcn_wmma_f32_16x16x32_f16(false, a1, false,
                  load_b_glb(w3t, 64, n * 16 + col16, 32, lane), (short)0, acc, false, false);
            const float bias = b3[n * 16 + col16];
            float mx = 0.f;   // safe: post-ReLU values are >=0 and >=1 row is valid
#pragma unroll
            for (int r = 0; r < 8; ++r) {
                const int grow = m0 + r + 8 * hi;
                float v = fmaxf(acc[r] + bias, 0.f);
                v = (grow < count) ? v : 0.f;
                mx = fmaxf(mx, v);
            }
            mx = fmaxf(mx, __shfl_xor(mx, 16, 32));   // fold the two 8-row halves
            if (lane < 16) pmax[w * FOUT + n * 16 + lane] = mx;
        }
    }
    __syncthreads();
    {
        const float m = fmaxf(fmaxf(pmax[tid], pmax[FOUT + tid]),
                              fmaxf(pmax[2 * FOUT + tid], pmax[3 * FOUT + tid]));
        out_x[(size_t)c * FOUT + tid] = m;
    }
}

// --------------------------- tail outputs -----------------------------------
__global__ __launch_bounds__(256)
void tail_kernel(const float* __restrict__ qpos, float* __restrict__ out) {
    const int i = blockIdx.x * blockDim.x + threadIdx.x;
    if (i >= BB * SS) return;
    float* out_pos = out + (size_t)BB * SS * FOUT;
    long long* out_batch = (long long*)(out + (size_t)BB * SS * FOUT + (size_t)BB * SS * 3);
    out_pos[i * 3 + 0] = qpos[i * 3 + 0];
    out_pos[i * 3 + 1] = qpos[i * 3 + 1];
    out_pos[i * 3 + 2] = qpos[i * 3 + 2];
    out_batch[i] = (long long)(i / SS);
}

// ----------------------------- launcher -------------------------------------
extern "C" void kernel_launch(void* const* d_in, const int* in_sizes, int n_in,
                              void* d_out, int out_size, void* d_ws, size_t ws_size,
                              hipStream_t stream) {
    const float* x   = (const float*)d_in[0];
    const float* pos = (const float*)d_in[1];
    // d_in[2] = batch (int64), unused: clouds are equal-sized
    const float* W1 = (const float*)d_in[3];
    const float* b1 = (const float*)d_in[4];
    const float* W2 = (const float*)d_in[5];
    const float* b2 = (const float*)d_in[6];
    const float* W3 = (const float*)d_in[7];
    const float* b3 = (const float*)d_in[8];
    float* out = (float*)d_out;
    char* ws = (char*)d_ws;

    // workspace partition (all 256B-aligned)
    int*   fps_idx = (int*)(ws + 0);                  // B*S ints        = 32768 B
    float* qpos    = (float*)(ws + 32768);            // B*S*3 floats    = 98304 B
    int*   nbr     = (int*)(ws + 131072);             // B*S*K ints      = 2 MiB
    int*   cnts    = (int*)(ws + 2228224);            // B*S ints        = 32768 B
    h16*   w1t     = (h16*)(ws + 2260992);            // 64*96 f16       = 12288 B
    h16*   w2t     = (h16*)(ws + 2273280);            // 64*64 f16       = 8192 B
    h16*   w3t     = (h16*)(ws + 2281472);            // 128*64 f16      = 16384 B

    wconv_kernel<<<1, 256, 0, stream>>>(W1, W2, W3, w1t, w2t, w3t);
    fps_kernel<<<BB, 1024, 0, stream>>>(pos, fps_idx, qpos);
    ballq_kernel<<<(BB * SS) / 8, 256, 0, stream>>>(pos, qpos, fps_idx, nbr, cnts);
    mlp_kernel<<<BB * SS, 128, 0, stream>>>(x, pos, nbr, cnts, qpos,
                                            w1t, w2t, w3t, b1, b2, b3, out);
    tail_kernel<<<(BB * SS + 255) / 256, 256, 0, stream>>>(qpos, out);
}
